// ShiftedWindowMSA_56916906606861
// MI455X (gfx1250) — matrix operand order repaired and checked by reading, not compiled, not run on hardware
//
#include <hip/hip_runtime.h>
#include <hip/hip_bf16.h>

// ---------------------------------------------------------------------------
// Shifted-window MSA fused into 3 WMMA GEMM stages (bf16 data, f32 accum)
// + one tiny weight-conversion prep kernel.
// B=32, grid 64x64, window 8x8 (M=64), 8 heads, head dim 16.
// All fragment loads are 16B-aligned b128; invariant fragments hoisted;
// all epilogues branch-free.
// ---------------------------------------------------------------------------

typedef __bf16 bf16;
typedef __attribute__((ext_vector_type(16))) __bf16 v16bf;
typedef __attribute__((ext_vector_type(8)))  float  v8f;

union Frag { v16bf v; unsigned u[8]; uint4 q[2]; };

__device__ __forceinline__ unsigned packbf(float a, float b) {
    union { unsigned u; __bf16 e[2]; } p;
    p.e[0] = (__bf16)a; p.e[1] = (__bf16)b;
    return p.u;
}

__device__ __forceinline__ uint4 and4(uint4 v, unsigned m) {
    return make_uint4(v.x & m, v.y & m, v.z & m, v.w & m);
}

// ---------------------------------------------------------------------------
// Kernel 0: one-time f32 -> bf16 conversion of w1 (384x128) and w2 (128x128).
// ---------------------------------------------------------------------------
__global__ __launch_bounds__(256) void cvt_kernel(
    const float* __restrict__ w1, const float* __restrict__ w2,
    bf16* __restrict__ w1b, bf16* __restrict__ w2b)
{
    const int i = blockIdx.x * 256 + threadIdx.x;
    if (i < 384 * 128) w1b[i] = (bf16)w1[i];
    if (i < 128 * 128) w2b[i] = (bf16)w2[i];
}

// ---------------------------------------------------------------------------
// Kernel 1: QKV projection fused with cyclic shift (-4,-4), window partition,
// head split. One block per (batch, window). QKV is one contiguous buffer:
//   k=0: Q[unit][m][e] (pre-scaled 1/sqrt(16));  k=1: K[unit][m][e];
//   k=2: Vt[unit][e][m] (transposed), each section 16M bf16.
// Wave w owns row-tile tm = w/2 and 12 col-tiles: A-fragments loaded once.
// ---------------------------------------------------------------------------
__global__ __launch_bounds__(256) void qkv_kernel(
    const float* __restrict__ x, const bf16* __restrict__ w1b,
    const float* __restrict__ b1, bf16* __restrict__ QKV)
{
    // row stride 136 bf16 = 68 dwords -> bank stride 4: conflict-free b128
    __shared__ bf16 xs[64 * 136];

    const int blk = blockIdx.x;            // b*64 + win
    const int b   = blk >> 6;
    const int win = blk & 63;
    const int Hw  = win >> 3, Ww = win & 7;

    // Stage the shift-gathered window tile (rolled source indexing)
    for (int idx = threadIdx.x; idx < 64 * 128; idx += 256) {
        const int m = idx >> 7, c = idx & 127;
        const int sy = (Hw * 8 + (m >> 3) + 4) & 63;
        const int sx = (Ww * 8 + (m & 7) + 4) & 63;
        xs[m * 136 + c] = (bf16)x[((size_t)b * 4096 + sy * 64 + sx) * 128 + c];
    }
    __syncthreads();

    const int wave = threadIdx.x >> 5;
    const int lane = threadIdx.x & 31;
    const int half = lane >> 4;
    const int l16  = lane & 15;

    const int tm  = wave >> 1;             // fixed row tile per wave
    const int tn0 = (wave & 1) * 12;

    // Preload all 4 A k-fragments for this wave's row tile
    Frag A[4];
    #pragma unroll
    for (int kt = 0; kt < 4; ++kt) {
        const bf16* ap = xs + (tm * 16 + l16) * 136 + kt * 32;
        A[kt].q[0] = *(const uint4*)(ap + 8 * half);       // K {0..7}+8h
        A[kt].q[1] = *(const uint4*)(ap + 16 + 8 * half);  // K {16..23}+8h
    }

    for (int t = 0; t < 12; ++t) {
        const int tn = tn0 + t;
        v8f acc = {};
        #pragma unroll
        for (int kt = 0; kt < 4; ++kt) {
            Frag B;
            const bf16* bp = w1b + (size_t)(tn * 16 + l16) * 128
                                 + kt * 32 + 8 * half;     // K {0..7}+8h, {16..23}+8h
            B.q[0] = *(const uint4*)bp;
            B.q[1] = *(const uint4*)(bp + 16);
            acc = __builtin_amdgcn_wmma_f32_16x16x32_bf16(
                false, A[kt].v, false, B.v, (short)0, acc, false, false);
        }
        // Branch-free epilogue: de-interleave j = (e*8+h)*3+k, scatter.
        const int j  = tn * 16 + l16;          // 0..383
        const int k  = j % 3;
        const int ch = j / 3;
        const int h  = ch & 7, e = ch >> 3;
        const float bias  = b1[j];
        const float scale = (k == 0) ? 0.25f : 1.0f;
        const int mbase = tm * 16 + 8 * half;
        const int str   = (k == 2) ? 1 : 16;   // Vt is [e][m]
        const int idx0  = (k == 2) ? (e * 64 + mbase) : (mbase * 16 + e);
        bf16* dst = QKV + (size_t)k * 16777216
                  + ((size_t)(b * 8 + h) * 64 + win) * 1024 + idx0;
        #pragma unroll
        for (int i = 0; i < 8; ++i)
            dst[i * str] = (bf16)((acc[i] + bias) * scale);
    }
}

// ---------------------------------------------------------------------------
// Kernel 2: windowed attention, one wave per (b, head, window) unit.
// S = Q*K^T (K-dim 16 zero-padded to 32) -> LDS; bias + shift masks +
// 3-pass softmax (2 rows/lane; LDS ops are in-order per wave); P packed bf16
// in place over the f32 rows; O = P*V; scatter to (B,L,C) bf16 activations.
// ---------------------------------------------------------------------------
__global__ __launch_bounds__(64) void attn_kernel(
    const bf16* __restrict__ Qb, const bf16* __restrict__ Kb,
    const bf16* __restrict__ Vb, const float* __restrict__ rel_bias,
    bf16* __restrict__ attn)
{
    __shared__ float S[2][64 * 68];        // 68-dword row stride, per wave
    __shared__ float rb[225];              // (15,15) relative bias table

    for (int i = threadIdx.x; i < 225; i += 64) rb[i] = rel_bias[i];
    __syncthreads();

    const int wave = threadIdx.x >> 5;
    const int lane = threadIdx.x & 31;
    const int half = lane >> 4;
    const int l16  = lane & 15;

    const int unit = blockIdx.x * 2 + wave;        // 0..16383
    const int win  = unit & 63;
    const int h    = (unit >> 6) & 7;
    const int b    = unit >> 9;
    const bf16* Q  = Qb + (size_t)unit * 1024;
    const bf16* K  = Kb + (size_t)unit * 1024;
    const bf16* V  = Vb + (size_t)unit * 1024;     // transposed: [e][m]
    float* Sw = S[wave];

    // Phase 1: S tiles = Q K^T  (16 WMMAs, K-dim zero-padded 16 -> 32).
    // K B-fragments hoisted; upper half zeroed by mask (no exec branching).
    const unsigned hm = half ? 0u : 0xFFFFFFFFu;
    Frag Bk[4];
    #pragma unroll
    for (int kt = 0; kt < 4; ++kt) {
        const bf16* kp = K + (kt * 16 + l16) * 16;
        Bk[kt].q[0] = *(const uint4*)(kp + 8 * half);
        Bk[kt].q[1] = and4(make_uint4(0u, 0u, 0u, 0u), hm);
    }
    for (int qt = 0; qt < 4; ++qt) {
        Frag A;
        A.q[0] = *(const uint4*)(Q + (qt * 16 + l16) * 16 + 8 * half);
        A.q[1] = make_uint4(0u, 0u, 0u, 0u);       // K >= 16 is padding
        #pragma unroll
        for (int kt = 0; kt < 4; ++kt) {
            v8f acc = {};
            acc = __builtin_amdgcn_wmma_f32_16x16x32_bf16(
                false, A.v, false, Bk[kt].v, (short)0, acc, false, false);
            #pragma unroll
            for (int i = 0; i < 8; ++i)
                Sw[(qt * 16 + i + 8 * half) * 68 + kt * 16 + l16] = acc[i];
        }
    }

    // Phase 2: bias + shift masks + softmax; rows lane and lane+32
    const bool lastRow = ((win >> 3) == 7);
    const bool lastCol = ((win & 7) == 7);
    for (int rr = 0; rr < 2; ++rr) {
        const int r  = lane + rr * 32;
        const int qh = r >> 3, qw = r & 7;
        float mx = -3.0e38f;
        for (int c = 0; c < 64; ++c) {
            float s = Sw[r * 68 + c];
            s += rb[(qh - (c >> 3) + 7) * 15 + (qw - (c & 7) + 7)];
            if (lastRow && ((r >= 32) != (c >= 32)))            s = -3.0e38f;
            if (lastCol && (((r & 7) >= 4) != ((c & 7) >= 4)))  s = -3.0e38f;
            Sw[r * 68 + c] = s;
            mx = fmaxf(mx, s);
        }
        float sum = 0.f;
        for (int c = 0; c < 64; ++c) {
            float e = __expf(Sw[r * 68 + c] - mx);
            sum += e;
            Sw[r * 68 + c] = e;
        }
        const float inv = 1.f / sum;
        for (int c = 0; c < 64; c += 2) {   // pack P bf16 over dwords 0..31
            float p0 = Sw[r * 68 + c]     * inv;
            float p1 = Sw[r * 68 + c + 1] * inv;
            Sw[r * 68 + (c >> 1)] = __uint_as_float(packbf(p0, p1));
        }
    }

    // Phase 3: O = P V (8 WMMAs); V B-fragments hoisted out of the ot loop.
    Frag Bv[2];
    #pragma unroll
    for (int kt = 0; kt < 2; ++kt) {
        const bf16* vp = V + l16 * 64 + kt * 32 + 8 * half;
        Bv[kt].q[0] = *(const uint4*)vp;
        Bv[kt].q[1] = *(const uint4*)(vp + 16);
    }
    for (int ot = 0; ot < 4; ++ot) {
        v8f acc = {};
        #pragma unroll
        for (int kt = 0; kt < 2; ++kt) {           // K = 64 = 2 x 32
            Frag A;
            // P dwords: pair k=2i+8h -> dword kt*16 + i + 4h (+8 for hi)
            const float* pp = Sw + (ot * 16 + l16) * 68 + kt * 16 + 4 * half;
            A.q[0] = *(const uint4*)pp;
            A.q[1] = *(const uint4*)(pp + 8);
            acc = __builtin_amdgcn_wmma_f32_16x16x32_bf16(
                false, A.v, false, Bv[kt].v, (short)0, acc, false, false);
        }
        #pragma unroll
        for (int i = 0; i < 8; ++i) {
            const int mtok = ot * 16 + i + 8 * half;
            const int dy = (win >> 3) * 8 + (mtok >> 3);
            const int dx = (win & 7) * 8 + (mtok & 7);
            const size_t l = (size_t)b * 4096 + dy * 64 + dx;
            attn[l * 128 + h * 16 + l16] = (bf16)acc[i];
        }
    }
}

// ---------------------------------------------------------------------------
// Kernel 3: output projection out = attn @ w2^T + b2, f32 result.
// One block = 128 rows x 128 cols; wave owns one row tile (A-fragments
// loaded once, reused for all 8 col-tiles).
// ---------------------------------------------------------------------------
__global__ __launch_bounds__(256) void proj_kernel(
    const bf16* __restrict__ attn, const bf16* __restrict__ w2b,
    const float* __restrict__ b2, float* __restrict__ out)
{
    const int wave = threadIdx.x >> 5;
    const int lane = threadIdx.x & 31;
    const int half = lane >> 4;
    const int l16  = lane & 15;
    const size_t row0 = (size_t)blockIdx.x * 128;
    const int tm = wave;

    Frag A[4];
    #pragma unroll
    for (int kt = 0; kt < 4; ++kt) {
        const bf16* ap = attn + (row0 + tm * 16 + l16) * 128
                              + kt * 32 + 8 * half;
        A[kt].q[0] = *(const uint4*)ap;
        A[kt].q[1] = *(const uint4*)(ap + 16);
    }

    for (int tn = 0; tn < 8; ++tn) {
        v8f acc = {};
        #pragma unroll
        for (int kt = 0; kt < 4; ++kt) {
            Frag B;
            const bf16* bp = w2b + (size_t)(tn * 16 + l16) * 128
                                 + kt * 32 + 8 * half;
            B.q[0] = *(const uint4*)bp;
            B.q[1] = *(const uint4*)(bp + 16);
            acc = __builtin_amdgcn_wmma_f32_16x16x32_bf16(
                false, A[kt].v, false, B.v, (short)0, acc, false, false);
        }
        const int j = tn * 16 + l16;
        const float bias = b2[j];
        #pragma unroll
        for (int i = 0; i < 8; ++i) {
            const int m = tm * 16 + i + 8 * half;
            out[(row0 + m) * 128 + j] = acc[i] + bias;
        }
    }
}

// ---------------------------------------------------------------------------
extern "C" void kernel_launch(void* const* d_in, const int* in_sizes, int n_in,
                              void* d_out, int out_size, void* d_ws, size_t ws_size,
                              hipStream_t stream)
{
    (void)in_sizes; (void)n_in; (void)out_size; (void)ws_size;
    const float* x   = (const float*)d_in[0];
    const float* w1  = (const float*)d_in[1];
    const float* b1  = (const float*)d_in[2];
    const float* w2  = (const float*)d_in[3];
    const float* b2  = (const float*)d_in[4];
    const float* rel = (const float*)d_in[5];

    // Workspace: Q | K | Vt | attn (each 16M bf16 = 32 MB) | w1b | w2b
    bf16* Qb  = (bf16*)d_ws;
    bf16* Kb  = Qb + (size_t)16777216;
    bf16* Vb  = Kb + (size_t)16777216;
    bf16* At  = Vb + (size_t)16777216;
    bf16* w1b = At + (size_t)16777216;
    bf16* w2b = w1b + (size_t)(384 * 128);

    cvt_kernel <<<192,  256, 0, stream>>>(w1, w2, w1b, w2b);
    qkv_kernel <<<2048, 256, 0, stream>>>(x, w1b, b1, Qb);
    attn_kernel<<<8192,  64, 0, stream>>>(Qb, Kb, Vb, rel, At);
    proj_kernel<<<1024, 256, 0, stream>>>(At, w2b, b2, (float*)d_out);
}
